// BiLSTM_CRF_63230508532020
// MI455X (gfx1250) — matrix operand order repaired and checked by reading, not compiled, not run on hardware
//
#include <hip/hip_runtime.h>

typedef unsigned short u16;
typedef unsigned int u32;
typedef __attribute__((ext_vector_type(4))) unsigned u32x4;
typedef __attribute__((ext_vector_type(4))) int v4i;
typedef __attribute__((ext_vector_type(8))) float v8f;
typedef __attribute__((ext_vector_type(16))) __bf16 v16bf;

#define B_ 64
#define L_ 512
#define EMB_ 256
#define HID_ 512
#define G4_ 2048
#define NT_ 6
#define PADTAG_ 5
#define NSPLIT 16

union Frag { u32x4 q[2]; v16bf v; };

__device__ __forceinline__ float bf2f(u16 h) { return __uint_as_float(((u32)h) << 16); }
__device__ __forceinline__ u16 f2bf(float f) {
  u32 u = __float_as_uint(f);
  u32 r = u + 0x7FFFu + ((u >> 16) & 1u);
  return (u16)(r >> 16);
}
__device__ __forceinline__ float sigmoidf_(float x) { return 1.0f / (1.0f + __expf(-x)); }
__device__ __forceinline__ v8f vzero8() { v8f v; for (int i = 0; i < 8; ++i) v[i] = 0.f; return v; }

// ---- CDNA5 async global->LDS (guarded; falls back to plain copy) ----
#if defined(__HIP_DEVICE_COMPILE__) && defined(__gfx1250__) && __has_builtin(__builtin_amdgcn_global_load_async_to_lds_b128)
#define HAVE_ASYNC 1
#else
#define HAVE_ASYNC 0
#endif

__device__ __forceinline__ void async_cp16(const void* g, void* l) {
#if HAVE_ASYNC
  __builtin_amdgcn_global_load_async_to_lds_b128(
      (__attribute__((address_space(1))) v4i*)(unsigned long long)(size_t)g,
      (__attribute__((address_space(3))) v4i*)(unsigned)(size_t)l, // flat shared addr: low 32 bits = LDS offset
      0, 0);
#else
  *(u32x4*)l = *(const u32x4*)g;
#endif
}
__device__ __forceinline__ void async_wait0() {
#if HAVE_ASYNC
#if __has_builtin(__builtin_amdgcn_s_wait_asynccnt)
  __builtin_amdgcn_s_wait_asynccnt(0);
#else
  asm volatile("s_wait_asynccnt 0" ::: "memory");
#endif
#endif
}

// ===================== prep kernels =====================
__global__ void cvt_bf16_k(const float* __restrict__ s, u16* __restrict__ d, int n) {
  int i = blockIdx.x * 256 + threadIdx.x;
  if (i < n) d[i] = f2bf(s[i]);
}
__global__ void bias_comb_k(const float* __restrict__ a, const float* __restrict__ b,
                            float* __restrict__ d, int n) {
  int i = blockIdx.x * 256 + threadIdx.x;
  if (i < n) d[i] = a[i] + b[i];
}
__global__ void zero_u32_k(u32* __restrict__ p, long n) {
  long i = (long)blockIdx.x * blockDim.x + threadIdx.x;
  long stride = (long)gridDim.x * blockDim.x;
  for (; i < n; i += stride) p[i] = 0u;
}
__global__ void embed_g_k(const int* __restrict__ text, const float* __restrict__ emb,
                          u16* __restrict__ xe) {
  int m = blockIdx.x;             // (b*L + t)
  int c = threadIdx.x;            // 0..255
  int tok = text[m];
  xe[(size_t)m * EMB_ + c] = f2bf(emb[(size_t)tok * EMB_ + c]);
}

// ===================== big input-projection GEMM (WMMA bf16) =====================
// C[m,n] (bf16) = sum_k A[m',k]*W[n,k] + bias[n], m' = optional per-row reversed index.
__global__ __launch_bounds__(256) void gemm_pre_k(
    const u16* __restrict__ A, int K,
    const u16* __restrict__ W,
    const float* __restrict__ bias,
    u16* __restrict__ C,
    int reverse, const int* __restrict__ lengths)
{
  __shared__ __align__(16) u16 As[128][40];
  __shared__ __align__(16) u16 Bs[128][40];
  const int tid = threadIdx.x;
  const int n0 = blockIdx.x * 128;
  const int m0 = blockIdx.y * 128;
  const int lrow = tid >> 1, half = tid & 1;

  int am = m0 + lrow;
  bool avalid = true;
  if (reverse) {
    int b = am >> 9, t = am & (L_ - 1);
    int len = lengths[b];
    avalid = (t < len);
    am = (b << 9) + (avalid ? (len - 1 - t) : 0);
  }
  const u16* aSrc = A + (size_t)am * K + half * 16;
  const u16* bSrc = W + (size_t)(n0 + lrow) * K + half * 16;

  const int lane = tid & 31;
  const int wv = tid >> 5;
  const int m_off = (wv >> 2) * 64;
  const int n_off = (wv & 3) * 32;
  const int r = lane & 15, kh = lane >> 4;

  v8f acc[4][2];
  for (int i = 0; i < 4; ++i) for (int j = 0; j < 2; ++j) acc[i][j] = vzero8();

  for (int kt = 0; kt < K; kt += 32) {
    u32x4 a0 = {0, 0, 0, 0}, a1 = {0, 0, 0, 0};
    if (avalid) {
      const u32x4* ap = (const u32x4*)(aSrc + kt);
      a0 = ap[0]; a1 = ap[1];
    }
    const u32x4* bp = (const u32x4*)(bSrc + kt);
    u32x4 b0 = bp[0], b1 = bp[1];
    *(u32x4*)&As[lrow][half * 16]     = a0;
    *(u32x4*)&As[lrow][half * 16 + 8] = a1;
    *(u32x4*)&Bs[lrow][half * 16]     = b0;
    *(u32x4*)&Bs[lrow][half * 16 + 8] = b1;
    if (kt + 32 < K) {                       // global_prefetch_b8 of next K-tile
      __builtin_prefetch(aSrc + kt + 32, 0, 1);
      __builtin_prefetch(bSrc + kt + 32, 0, 1);
    }
    __syncthreads();
    Frag aF[4], bF[2];
    for (int mi = 0; mi < 4; ++mi) {         // A 16x32: chunks at K = 8*(lane>=16), +16
      const u16* p = &As[m_off + mi * 16 + r][kh * 8];
      aF[mi].q[0] = *(const u32x4*)p;
      aF[mi].q[1] = *(const u32x4*)(p + 16);
    }
    for (int ni = 0; ni < 2; ++ni) {         // B 32x16: chunks at K = 8*(lane>=16), +16
      const u16* p = &Bs[n_off + ni * 16 + r][kh * 8];
      bF[ni].q[0] = *(const u32x4*)p;
      bF[ni].q[1] = *(const u32x4*)(p + 16);
    }
    for (int mi = 0; mi < 4; ++mi)
      for (int ni = 0; ni < 2; ++ni)
        acc[mi][ni] = __builtin_amdgcn_wmma_f32_16x16x32_bf16(
            false, aF[mi].v, false, bF[ni].v, (short)0, acc[mi][ni], false, false);
    __syncthreads();
  }
  for (int mi = 0; mi < 4; ++mi)
    for (int ni = 0; ni < 2; ++ni) {
      int gn = n0 + n_off + ni * 16 + (lane & 15);
      float bs = bias ? bias[gn] : 0.f;
      int rowbase = m0 + m_off + mi * 16 + ((lane >> 4) << 3);
      for (int k = 0; k < 8; ++k)
        C[(size_t)(rowbase + k) * G4_ + gn] = f2bf(acc[mi][ni][k] + bs);
    }
}

// ===================== persistent recurrent kernel =====================
// grid (NSPLIT, 4 batch-groups, 2 dirs), 64 threads (2 waves).
// Whh slice (128 gate-rows x 512) pinned in LDS for all 512 steps.
__global__ __launch_bounds__(64) void lstm_seq_k(
    const u16* __restrict__ pre,             // [2][B][L][4H] bf16 (bias folded)
    const u16* __restrict__ whhF, const u16* __restrict__ whhB,
    u16* __restrict__ out,                   // [B][L][2H] bf16 (zero-initialized)
    u16* __restrict__ hbuf,                  // [2 parity][2 dir][4 bg][16][H]
    int* __restrict__ bar,
    const int* __restrict__ lengths,
    int barBase)
{
  __shared__ __align__(16) u16 Wsh[128][520];
  __shared__ __align__(16) u16 Hsh[16][520];
  const int ns = blockIdx.x, bg = blockIdx.y, dir = blockIdx.z;
  const int tid = threadIdx.x, lane = tid & 31, wv = tid >> 5;
  const u16* whh = dir ? whhB : whhF;
  const int rb0 = bg * 16;
  const int hc0 = ns * 32 + wv * 16;         // this wave's h-column base

  // one-time Whh slice -> LDS (async where available)
  for (int idx = tid; idx < 128 * 64; idx += 64) {
    int lr = idx >> 6, c = idx & 63;
    int g = lr >> 5, j = lr & 31;
    async_cp16(whh + ((size_t)(g * HID_ + ns * 32 + j)) * HID_ + c * 8, &Wsh[lr][c * 8]);
  }
  for (int idx = tid; idx < 16 * 65; idx += 64) {   // h_0 = 0
    u32x4 z = {0, 0, 0, 0};
    *(u32x4*)&Hsh[idx / 65][(idx % 65) * 8] = z;
  }
  async_wait0();
  __syncthreads();

  const int cc = lane & 15;
  const int rbase = (lane >> 4) << 3;         // C-layout: row = vgpr + 8*(lane>=16)
  int lenv[8];
  for (int k = 0; k < 8; ++k) lenv[k] = lengths[rb0 + rbase + k];
  float c_st[8];
  for (int k = 0; k < 8; ++k) c_st[k] = 0.f;

  int* barp = bar + barBase + dir * 4 + bg;

  for (int t = 0; t < L_; ++t) {
    v8f acc[4];
    for (int g = 0; g < 4; ++g) {             // init accumulators from hoisted pre + bias
      v8f a;
      for (int k = 0; k < 8; ++k)
        a[k] = bf2f(pre[((size_t)dir * B_ * L_ + (size_t)(rb0 + rbase + k) * L_ + t) * G4_ +
                        g * HID_ + hc0 + cc]);
      acc[g] = a;
    }
    for (int kt = 0; kt < HID_; kt += 32) {   // 16x512x(4x16) GEMM step
      Frag aF;
      const u16* ap = &Hsh[lane & 15][kt + (lane >> 4) * 8];
      aF.q[0] = *(const u32x4*)ap;
      aF.q[1] = *(const u32x4*)(ap + 16);
      for (int g = 0; g < 4; ++g) {
        Frag bF;
        const u16* bp = &Wsh[g * 32 + wv * 16 + cc][kt + (lane >> 4) * 8];
        bF.q[0] = *(const u32x4*)bp;
        bF.q[1] = *(const u32x4*)(bp + 16);
        acc[g] = __builtin_amdgcn_wmma_f32_16x16x32_bf16(
            false, aF.v, false, bF.v, (short)0, acc[g], false, false);
      }
    }
    // i/f/g/o live in the same lane/slot -> pure per-thread cell update
    u16* hslot = hbuf + ((((size_t)(t & 1) * 2 + dir) * 4 + bg) * 16) * HID_;
    for (int k = 0; k < 8; ++k) {
      float ig = sigmoidf_(acc[0][k]);
      float fg = sigmoidf_(acc[1][k]);
      float gg = tanhf(acc[2][k]);
      float og = sigmoidf_(acc[3][k]);
      c_st[k] = fg * c_st[k] + ig * gg;
      float hv = og * tanhf(c_st[k]);
      u16 hb = f2bf(hv);
      int row = rbase + k;
      hslot[row * HID_ + hc0 + cc] = hb;
      int len = lenv[k];
      if (t < len) {
        int tout = dir ? (len - 1 - t) : t;
        out[((size_t)(rb0 + row) * L_ + tout) * (2 * HID_) + dir * HID_ + hc0 + cc] = hb;
      }
    }
    __syncthreads();
    if (tid == 0) {                           // device-scope arrive/spin across the 16 WGs
      __threadfence();
      atomicAdd(barp, 1);
      int target = NSPLIT * (t + 1);
      while (atomicAdd(barp, 0) < target) __builtin_amdgcn_s_sleep(2);
      __threadfence();
    }
    __syncthreads();
    const u16* hsrc = hbuf + ((((size_t)(t & 1) * 2 + dir) * 4 + bg) * 16) * HID_;
    for (int idx = tid; idx < 16 * 64; idx += 64) {
      int rr = idx >> 6, c = idx & 63;
      async_cp16(hsrc + rr * HID_ + c * 8, &Hsh[rr][c * 8]);
    }
    async_wait0();
    __syncthreads();
  }
}

// ===================== FC + softmax =====================
__global__ __launch_bounds__(256) void fc_softmax_k(
    const u16* __restrict__ x, const u16* __restrict__ fw,
    const float* __restrict__ fb, float* __restrict__ probs)
{
  __shared__ u16 Wl[NT_ * 1024];
  for (int i = threadIdx.x; i < NT_ * 1024; i += 256) Wl[i] = fw[i];
  __syncthreads();
  int wv = threadIdx.x >> 5, lane = threadIdx.x & 31;
  int m = blockIdx.x * 8 + wv;
  const u16* row = x + (size_t)m * 1024;
  float p[NT_];
  for (int t = 0; t < NT_; ++t) p[t] = 0.f;
  for (int j = 0; j < 32; ++j) {
    int idx = j * 32 + lane;
    float xv = bf2f(row[idx]);
    for (int t = 0; t < NT_; ++t) p[t] += xv * bf2f(Wl[t * 1024 + idx]);
  }
  for (int t = 0; t < NT_; ++t)
    for (int off = 16; off; off >>= 1) p[t] += __shfl_xor(p[t], off, 32);
  if (lane == 0) {
    float l[NT_], mx = -1e30f;
    for (int t = 0; t < NT_; ++t) { l[t] = p[t] + fb[t]; mx = fmaxf(mx, l[t]); }
    float s = 0.f;
    for (int t = 0; t < NT_; ++t) { l[t] = __expf(l[t] - mx); s += l[t]; }
    float inv = 1.f / s;
    for (int t = 0; t < NT_; ++t) probs[(size_t)m * NT_ + t] = l[t] * inv;
  }
}

// ===================== Viterbi (T=6) =====================
__global__ __launch_bounds__(32) void viterbi_k(
    const float* __restrict__ probs, const int* __restrict__ lengths,
    const float* __restrict__ cs, const float* __restrict__ ce,
    const float* __restrict__ ctr, int* __restrict__ outTags)
{
  __shared__ float score[NT_];
  __shared__ unsigned char hist[L_][NT_];
  __shared__ float tr[NT_ * NT_];
  int b = blockIdx.x, tid = threadIdx.x;
  for (int i = tid; i < NT_ * NT_; i += 32) tr[i] = ctr[i];
  int len = lengths[b];
  const float* em = probs + (size_t)b * L_ * NT_;
  if (tid < NT_) { score[tid] = cs[tid] + em[tid]; hist[L_ - 1][tid] = 0; }
  __syncthreads();
  for (int t = 1; t < L_; ++t) {
    float ns = -1e30f; int bi = 0;
    if (tid < NT_) {
      ns = score[0] + tr[tid];
      for (int i = 1; i < NT_; ++i) {
        float v = score[i] + tr[i * NT_ + tid];
        if (v > ns) { ns = v; bi = i; }
      }
    }
    __syncthreads();
    if (tid < NT_) {
      hist[t - 1][tid] = (unsigned char)bi;
      if (t < len) score[tid] = ns + em[t * NT_ + tid];
    }
    __syncthreads();
  }
  if (tid == 0) {
    int bl = 0; float bv = score[0] + ce[0];
    for (int j = 1; j < NT_; ++j) {
      float v = score[j] + ce[j];
      if (v > bv) { bv = v; bl = j; }
    }
    int end = len - 1, carry = 0;
    for (int t = L_ - 1; t >= 0; --t) {
      int back = hist[t][carry];
      int tag = (t == end) ? bl : back;
      outTags[(size_t)b * L_ + t] = (t <= end) ? tag : PADTAG_;
      carry = tag;
    }
  }
}

// ===================== host orchestration =====================
extern "C" void kernel_launch(void* const* d_in, const int* in_sizes, int n_in,
                              void* d_out, int out_size, void* d_ws, size_t ws_size,
                              hipStream_t stream) {
  (void)in_sizes; (void)n_in; (void)out_size; (void)ws_size;
  const int* text     = (const int*)d_in[0];
  const int* lengths  = (const int*)d_in[1];
  const float* embed  = (const float*)d_in[3];
  const float* wih[4] = {(const float*)d_in[4],  (const float*)d_in[8],
                         (const float*)d_in[12], (const float*)d_in[16]};
  const float* whh[4] = {(const float*)d_in[5],  (const float*)d_in[9],
                         (const float*)d_in[13], (const float*)d_in[17]};
  const float* bih[4] = {(const float*)d_in[6],  (const float*)d_in[10],
                         (const float*)d_in[14], (const float*)d_in[18]};
  const float* bhh[4] = {(const float*)d_in[7],  (const float*)d_in[11],
                         (const float*)d_in[15], (const float*)d_in[19]};
  const float* fcw = (const float*)d_in[20];
  const float* fcb = (const float*)d_in[21];
  const float* crf_s = (const float*)d_in[22];
  const float* crf_e = (const float*)d_in[23];
  const float* crf_t = (const float*)d_in[24];

  char* ws = (char*)d_ws;
  size_t off = 0;
  auto alloc = [&](size_t bytes) -> void* {
    void* p = ws + off;
    off = (off + bytes + 255) & ~(size_t)255;
    return p;
  };
  u16* bWIH0F = (u16*)alloc((size_t)G4_ * EMB_ * 2);
  u16* bWIH0B = (u16*)alloc((size_t)G4_ * EMB_ * 2);
  u16* bWIH1F = (u16*)alloc((size_t)G4_ * 1024 * 2);
  u16* bWIH1B = (u16*)alloc((size_t)G4_ * 1024 * 2);
  u16* bWHH[4];
  for (int i = 0; i < 4; ++i) bWHH[i] = (u16*)alloc((size_t)G4_ * HID_ * 2);
  u16* bFCW  = (u16*)alloc((size_t)NT_ * 1024 * 2);
  float* fBIAS = (float*)alloc((size_t)4 * G4_ * 4);
  u16* XE   = (u16*)alloc((size_t)B_ * L_ * EMB_ * 2);
  u16* PRE  = (u16*)alloc((size_t)2 * B_ * L_ * G4_ * 2);
  char* zstart = ws + off;
  u16* OUT0 = (u16*)alloc((size_t)B_ * L_ * 1024 * 2);
  u16* OUT1 = (u16*)alloc((size_t)B_ * L_ * 1024 * 2);
  u16* HBUF = (u16*)alloc((size_t)2 * 2 * 4 * 16 * HID_ * 2);
  int* BAR  = (int*)alloc(64 * 4);
  long zcount = (long)((ws + off) - zstart) / 4;
  float* PROBS = (float*)alloc((size_t)B_ * L_ * NT_ * 4);

  auto cvt = [&](const float* s, u16* d, int n) {
    cvt_bf16_k<<<(n + 255) / 256, 256, 0, stream>>>(s, d, n);
  };
  cvt(wih[0], bWIH0F, G4_ * EMB_);
  cvt(wih[1], bWIH0B, G4_ * EMB_);
  cvt(wih[2], bWIH1F, G4_ * 1024);
  cvt(wih[3], bWIH1B, G4_ * 1024);
  for (int i = 0; i < 4; ++i) cvt(whh[i], bWHH[i], G4_ * HID_);
  cvt(fcw, bFCW, NT_ * 1024);
  for (int i = 0; i < 4; ++i)
    bias_comb_k<<<G4_ / 256, 256, 0, stream>>>(bih[i], bhh[i], fBIAS + i * G4_, G4_);

  zero_u32_k<<<4096, 256, 0, stream>>>((u32*)zstart, zcount);
  embed_g_k<<<B_ * L_, EMB_, 0, stream>>>(text, embed, XE);

  u16* PRE1 = PRE + (size_t)B_ * L_ * G4_;
  dim3 gG(G4_ / 128, (B_ * L_) / 128);

  // layer 0
  gemm_pre_k<<<gG, 256, 0, stream>>>(XE, EMB_, bWIH0F, fBIAS + 0 * G4_, PRE,  0, lengths);
  gemm_pre_k<<<gG, 256, 0, stream>>>(XE, EMB_, bWIH0B, fBIAS + 1 * G4_, PRE1, 1, lengths);
  lstm_seq_k<<<dim3(NSPLIT, 4, 2), 64, 0, stream>>>(PRE, bWHH[0], bWHH[1], OUT0, HBUF, BAR,
                                                    lengths, 0);
  // layer 1
  gemm_pre_k<<<gG, 256, 0, stream>>>(OUT0, 1024, bWIH1F, fBIAS + 2 * G4_, PRE,  0, lengths);
  gemm_pre_k<<<gG, 256, 0, stream>>>(OUT0, 1024, bWIH1B, fBIAS + 3 * G4_, PRE1, 1, lengths);
  lstm_seq_k<<<dim3(NSPLIT, 4, 2), 64, 0, stream>>>(PRE, bWHH[2], bWHH[3], OUT1, HBUF, BAR,
                                                    lengths, 8);
  // head
  fc_softmax_k<<<(B_ * L_) / 8, 256, 0, stream>>>(OUT1, bFCW, fcb, PROBS);
  viterbi_k<<<B_, 32, 0, stream>>>(PROBS, lengths, crf_s, crf_e, crf_t, (int*)d_out);
}
